// Attention_9861244912306
// MI455X (gfx1250) — matrix operand order repaired and checked by reading, not compiled, not run on hardware
//
#include <hip/hip_runtime.h>

// ---------------------------------------------------------------------------
// Types / helpers
// ---------------------------------------------------------------------------
typedef float  v8f   __attribute__((ext_vector_type(8)));
typedef __bf16 v16bf __attribute__((ext_vector_type(16)));
typedef __bf16 v8bf  __attribute__((ext_vector_type(8)));

union Frag16 { v16bf v; v8bf h[2]; };

#if defined(__has_builtin)
#if __has_builtin(__builtin_amdgcn_cvt_pk_bf16_f32)
#define HAVE_PK_BF16 1
#endif
#if __has_builtin(__builtin_amdgcn_exp2f)
#define HAVE_EXP2 1
#endif
#endif

// pack two f32 -> two bf16 (one v_cvt_pk_bf16_f32 when available)
__device__ __forceinline__ unsigned pk2(float a, float b) {
#ifdef HAVE_PK_BF16
  auto p = __builtin_amdgcn_cvt_pk_bf16_f32(a, b);
  return __builtin_bit_cast(unsigned, p);
#else
  unsigned ua = __builtin_bit_cast(unsigned, a);
  ua += 0x7FFFu + ((ua >> 16) & 1u);
  unsigned ub = __builtin_bit_cast(unsigned, b);
  ub += 0x7FFFu + ((ub >> 16) & 1u);
  return (ua >> 16) | (ub & 0xFFFF0000u);
#endif
}

__device__ __forceinline__ __bf16 f2bf(float f) {
  unsigned u = pk2(f, f);
  return __builtin_bit_cast(__bf16, (unsigned short)u);
}

__device__ __forceinline__ float fast_exp2(float x) {
#ifdef HAVE_EXP2
  return __builtin_amdgcn_exp2f(x);
#else
  return exp2f(x);
#endif
}

__device__ __forceinline__ v8f wmma_bf16(v16bf a, v16bf b, v8f c) {
  return __builtin_amdgcn_wmma_f32_16x16x32_bf16(false, a, false, b, (short)0, c,
                                                 false, false);
}

// DPP ROW_XMASK lane^mask move within the 16-lane row (pure VALU, no LDS)
template <int MASK>
__device__ __forceinline__ float dpp_xmask(float x) {
  int i = __builtin_bit_cast(int, x);
  int y = __builtin_amdgcn_update_dpp(0, i, 0x160 | MASK, 0xF, 0xF, true);
  return __builtin_bit_cast(float, y);
}

__device__ __forceinline__ float rowmax16(float x) {
  x = fmaxf(x, dpp_xmask<1>(x));
  x = fmaxf(x, dpp_xmask<2>(x));
  x = fmaxf(x, dpp_xmask<4>(x));
  x = fmaxf(x, dpp_xmask<8>(x));
  return x;
}

// ---------------------------------------------------------------------------
// Async global -> LDS copy (CDNA5 ASYNCcnt path); 16B per lane.
// VDST = LDS byte offset (low 32 bits of generic pointer to __shared__),
// VADDR = 64-bit global address, no SADDR.
// ---------------------------------------------------------------------------
__device__ __forceinline__ void async_g2l_b128(void* lds, const void* g) {
  unsigned loff = (unsigned)(unsigned long long)lds;
  asm volatile("global_load_async_to_lds_b128 %0, %1, off"
               :
               : "v"(loff), "v"(g)
               : "memory");
}
__device__ __forceinline__ void wait_async0() {
  asm volatile("s_wait_asynccnt 0x0" ::: "memory");
}

// ---------------------------------------------------------------------------
// Bulk f32 -> bf16 conversion (8 elements / thread), bandwidth-bound
// ---------------------------------------------------------------------------
__global__ __launch_bounds__(256) void cvt_bf16_kernel(
    const float* __restrict__ src, __bf16* __restrict__ dst, int n8) {
  const int i = blockIdx.x * 256 + threadIdx.x;
  if (i >= n8) return;
  const float4* s = (const float4*)src + (size_t)i * 2;
  float4 a = s[0], b = s[1];
  union { unsigned u[4]; uint4 q; } t;
  t.u[0] = pk2(a.x, a.y);
  t.u[1] = pk2(a.z, a.w);
  t.u[2] = pk2(b.x, b.y);
  t.u[3] = pk2(b.z, b.w);
  *(uint4*)(dst + (size_t)i * 8) = t.q;
}

// ---------------------------------------------------------------------------
// WMMA GEMM (all-bf16 operands):  Y[m][n] = sum_k X[m][k] * W[n][k]
// bf16 out: * oscale ; f32 out: + bias[n].
// Block tile 128x128, K-chunk 32, 8 waves (2x4), wave tile 64x32.
// Tiles staged with async global->LDS copies.
// ---------------------------------------------------------------------------
template <bool OUT_F32>
__global__ __launch_bounds__(256) void gemm_wmma_kernel(
    const __bf16* __restrict__ X, const __bf16* __restrict__ W,
    const float* __restrict__ bias, void* __restrict__ Yv,
    int M, int K, int N, float oscale) {
  __shared__ __attribute__((aligned(64))) __bf16 As[128 * 32];
  __shared__ __attribute__((aligned(64))) __bf16 Bs[128 * 32];

  const int tid  = threadIdx.x;
  const int lane = tid & 31;
  const int wv   = tid >> 5;
  const int wm   = wv >> 2;
  const int wn   = wv & 3;
  const int half = lane >> 4;
  const int l16  = lane & 15;

  const int mBase = blockIdx.x * 128;
  const int nBase = blockIdx.y * 128;

  v8f zero = {0, 0, 0, 0, 0, 0, 0, 0};
  v8f acc[4][2];
#pragma unroll
  for (int i = 0; i < 4; i++)
#pragma unroll
    for (int j = 0; j < 2; j++) acc[i][j] = zero;

  const int ldRow = tid >> 1;
  const int ldK   = (tid & 1) * 16;

  for (int kc = 0; kc < K; kc += 32) {
    {
      const __bf16* px = X + (size_t)(mBase + ldRow) * K + kc + ldK;
      async_g2l_b128(&As[ldRow * 32 + ldK], px);
      async_g2l_b128(&As[ldRow * 32 + ldK + 8], px + 8);
      const __bf16* pw = W + (size_t)(nBase + ldRow) * K + kc + ldK;
      async_g2l_b128(&Bs[ldRow * 32 + ldK], pw);
      async_g2l_b128(&Bs[ldRow * 32 + ldK + 8], pw + 8);
    }
    wait_async0();
    __syncthreads();

    Frag16 bf0, bf1;
    bf0.v = *(const v16bf*)&Bs[(wn * 32 + l16) * 32 + half * 16];
    bf1.v = *(const v16bf*)&Bs[(wn * 32 + 16 + l16) * 32 + half * 16];

#pragma unroll
    for (int mi = 0; mi < 4; mi++) {
      const int ml = wm * 64 + mi * 16 + l16;
      Frag16 a;
      a.h[0] = *(const v8bf*)&As[ml * 32 + half * 16];
      a.h[1] = *(const v8bf*)&As[ml * 32 + half * 16 + 8];
      acc[mi][0] = wmma_bf16(a.v, bf0.v, acc[mi][0]);
      acc[mi][1] = wmma_bf16(a.v, bf1.v, acc[mi][1]);
    }
    __syncthreads();
  }

#pragma unroll
  for (int mi = 0; mi < 4; mi++)
#pragma unroll
    for (int ni = 0; ni < 2; ni++)
#pragma unroll
      for (int r = 0; r < 8; r++) {
        const int m = mBase + wm * 64 + mi * 16 + half * 8 + r;
        const int n = nBase + wn * 32 + ni * 16 + l16;
        const float v = acc[mi][ni][r];
        if constexpr (OUT_F32) {
          ((float*)Yv)[(size_t)m * N + n] = v + bias[n];
        } else {
          ((__bf16*)Yv)[(size_t)m * N + n] = f2bf(v * oscale);
        }
      }
}

// ---------------------------------------------------------------------------
// Flash attention over bf16 Q/K/V in [b][seq][h*64+d] layout.
// Q pre-scaled by (1/sqrt(64))*log2(e); softmax in base-2 (v_exp_f32 native).
// Block = 128 queries of one (b,h); 8 waves x 16 queries; 64-key tiles.
// Q/K tiles staged with async global->LDS; V transposed through VGPRs.
// Row max via DPP xmask; row sum via WMMA against a ones matrix.
// ---------------------------------------------------------------------------
__global__ __launch_bounds__(256) void attn_kernel(
    const __bf16* __restrict__ Qb, const __bf16* __restrict__ Kb,
    const __bf16* __restrict__ Vb, __bf16* __restrict__ Ob) {
  __shared__ __attribute__((aligned(64))) __bf16 Qs[128 * 64];
  __shared__ __attribute__((aligned(64))) __bf16 Ks[64 * 64];
  __shared__ __attribute__((aligned(64))) __bf16 Vt[64 * 64];
  __shared__ __attribute__((aligned(64))) __bf16 Ps[8][16 * 64];

  const int tid  = threadIdx.x;
  const int lane = tid & 31;
  const int w    = tid >> 5;
  const int half = lane >> 4;
  const int l16  = lane & 15;

  const int bx   = blockIdx.x;
  const int qblk = bx & 15;
  const int h    = (bx >> 4) & 7;
  const int b    = bx >> 7;
  const int q0   = qblk * 128;

  {
    const int row = tid >> 1;
    const int cs  = (tid & 1) * 32;
    const __bf16* src = Qb + ((size_t)(b * 2048 + q0 + row)) * 512 + h * 64 + cs;
    async_g2l_b128(&Qs[row * 64 + cs], src);
    async_g2l_b128(&Qs[row * 64 + cs + 8], src + 8);
    async_g2l_b128(&Qs[row * 64 + cs + 16], src + 16);
    async_g2l_b128(&Qs[row * 64 + cs + 24], src + 24);
  }
  wait_async0();
  __syncthreads();

  Frag16 qf[2];
  {
    const int ml = w * 16 + l16;
#pragma unroll
    for (int c = 0; c < 2; c++) {
      qf[c].h[0] = *(const v8bf*)&Qs[ml * 64 + c * 32 + half * 16];
      qf[c].h[1] = *(const v8bf*)&Qs[ml * 64 + c * 32 + half * 16 + 8];
    }
  }

  float run_m[8], run_l[8];
#pragma unroll
  for (int r = 0; r < 8; r++) { run_m[r] = -3.0e38f; run_l[r] = 0.0f; }
  v8f zero = {0, 0, 0, 0, 0, 0, 0, 0};
  v8f o[4];
#pragma unroll
  for (int d = 0; d < 4; d++) o[d] = zero;

  v16bf ones;
#pragma unroll
  for (int i = 0; i < 16; i++) ones[i] = (__bf16)1.0f;

  const int key = tid >> 2;
  const int d16 = (tid & 3) * 16;
  const size_t kvBase = ((size_t)b * 4096) * 512 + h * 64 + d16;

  for (int kt = 0; kt < 64; kt++) {
    {
      const size_t off = kvBase + (size_t)(kt * 64 + key) * 512;
      // K tile: async direct copy (row-major)
      async_g2l_b128(&Ks[key * 64 + d16], Kb + off);
      async_g2l_b128(&Ks[key * 64 + d16 + 8], Kb + off + 8);
      // V tile: transpose through VGPRs
      v8bf v0 = *(const v8bf*)(Vb + off);
      v8bf v1 = *(const v8bf*)(Vb + off + 8);
#pragma unroll
      for (int i = 0; i < 8; i++) Vt[(d16 + i) * 64 + key] = v0[i];
#pragma unroll
      for (int i = 0; i < 8; i++) Vt[(d16 + 8 + i) * 64 + key] = v1[i];
      if (kt < 63) {
        __builtin_prefetch(Kb + off + (size_t)64 * 512, 0, 0);
        __builtin_prefetch(Vb + off + (size_t)64 * 512, 0, 0);
      }
    }
    wait_async0();
    __syncthreads();

    v8f s[4];
#pragma unroll
    for (int j = 0; j < 4; j++) s[j] = zero;
#pragma unroll
    for (int c = 0; c < 2; c++)
#pragma unroll
      for (int j = 0; j < 4; j++) {
        v16bf bk = *(const v16bf*)&Ks[(j * 16 + l16) * 64 + c * 32 + half * 16];
        s[j] = wmma_bf16(qf[c].v, bk, s[j]);
      }

    __bf16* P = &Ps[w][0];
    float cf[8];
#pragma unroll
    for (int r = 0; r < 8; r++) {
      float mx = fmaxf(fmaxf(s[0][r], s[1][r]), fmaxf(s[2][r], s[3][r]));
      mx = rowmax16(mx);
      float nm = fmaxf(run_m[r], mx);
      cf[r] = fast_exp2(run_m[r] - nm);   // base-2 softmax (log2e folded in Q)
      run_m[r] = nm;
      const int mrow = half * 8 + r;
#pragma unroll
      for (int j = 0; j < 4; j++) {
        float p = fast_exp2(s[j][r] - nm);
        P[mrow * 64 + j * 16 + l16] = f2bf(p);
      }
    }

    Frag16 pf0, pf1;
    pf0.h[0] = *(const v8bf*)&P[l16 * 64 + half * 16];
    pf0.h[1] = *(const v8bf*)&P[l16 * 64 + half * 16 + 8];
    pf1.h[0] = *(const v8bf*)&P[l16 * 64 + 32 + half * 16];
    pf1.h[1] = *(const v8bf*)&P[l16 * 64 + 32 + half * 16 + 8];

    v8f lsum = wmma_bf16(pf0.v, ones, zero);
    lsum = wmma_bf16(pf1.v, ones, lsum);
#pragma unroll
    for (int r = 0; r < 8; r++) run_l[r] = run_l[r] * cf[r] + lsum[r];

#pragma unroll
    for (int d = 0; d < 4; d++)
#pragma unroll
      for (int r = 0; r < 8; r++) o[d][r] *= cf[r];
#pragma unroll
    for (int d = 0; d < 4; d++) {
      v16bf bv0 = *(const v16bf*)&Vt[(d * 16 + l16) * 64 + half * 16];
      v16bf bv1 = *(const v16bf*)&Vt[(d * 16 + l16) * 64 + 32 + half * 16];
      o[d] = wmma_bf16(pf0.v, bv0, o[d]);
      o[d] = wmma_bf16(pf1.v, bv1, o[d]);
    }
    __syncthreads();
  }

  float inv[8];
#pragma unroll
  for (int r = 0; r < 8; r++) inv[r] = 1.0f / run_l[r];
#pragma unroll
  for (int d = 0; d < 4; d++)
#pragma unroll
    for (int r = 0; r < 8; r++) {
      const int q = q0 + w * 16 + half * 8 + r;
      Ob[((size_t)(b * 2048 + q)) * 512 + h * 64 + d * 16 + l16] =
          f2bf(o[d][r] * inv[r]);
    }
}

// ---------------------------------------------------------------------------
// Launch
// ---------------------------------------------------------------------------
extern "C" void kernel_launch(void* const* d_in, const int* in_sizes, int n_in,
                              void* d_out, int out_size, void* d_ws, size_t ws_size,
                              hipStream_t stream) {
  (void)in_sizes; (void)n_in; (void)out_size; (void)ws_size;
  const float* q   = (const float*)d_in[0];
  const float* kv  = (const float*)d_in[1];
  const float* w_q = (const float*)d_in[2];
  const float* w_k = (const float*)d_in[3];
  const float* w_v = (const float*)d_in[4];
  const float* w_o = (const float*)d_in[5];
  const float* b_o = (const float*)d_in[6];

  // bf16 workspace layout
  __bf16* qbf  = (__bf16*)d_ws;                    // 8192 x 512
  __bf16* kvbf = qbf  + (size_t)8192 * 512;        // 16384 x 128
  __bf16* wqb  = kvbf + (size_t)16384 * 128;       // 512 x 512
  __bf16* wkb  = wqb  + (size_t)512 * 512;         // 512 x 128
  __bf16* wvb  = wkb  + (size_t)512 * 128;         // 512 x 128
  __bf16* wob  = wvb  + (size_t)512 * 128;         // 512 x 512
  __bf16* Qb   = wob  + (size_t)512 * 512;         // 8192 x 512
  __bf16* Kb   = Qb   + (size_t)8192 * 512;        // 16384 x 512
  __bf16* Vb   = Kb   + (size_t)16384 * 512;       // 16384 x 512
  __bf16* Ob   = Vb   + (size_t)16384 * 512;       // 8192 x 512

  dim3 blk(256);
  // one-time f32 -> bf16 conversions (weights converted once, not per block)
  cvt_bf16_kernel<<<dim3(2048), blk, 0, stream>>>(q,   qbf,  8192 * 512 / 8);
  cvt_bf16_kernel<<<dim3(1024), blk, 0, stream>>>(kv,  kvbf, 16384 * 128 / 8);
  cvt_bf16_kernel<<<dim3(128),  blk, 0, stream>>>(w_q, wqb,  512 * 512 / 8);
  cvt_bf16_kernel<<<dim3(32),   blk, 0, stream>>>(w_k, wkb,  512 * 128 / 8);
  cvt_bf16_kernel<<<dim3(32),   blk, 0, stream>>>(w_v, wvb,  512 * 128 / 8);
  cvt_bf16_kernel<<<dim3(128),  blk, 0, stream>>>(w_o, wob,  512 * 512 / 8);

  // Q projection: fold 1/sqrt(64) * log2(e) into bf16 epilogue (base-2 softmax)
  const float qscale = 0.125f * 1.44269504088896340736f;
  gemm_wmma_kernel<false><<<dim3(64, 4), blk, 0, stream>>>(
      qbf, wqb, nullptr, Qb, 8192, 512, 512, qscale);
  gemm_wmma_kernel<false><<<dim3(128, 4), blk, 0, stream>>>(
      kvbf, wkb, nullptr, Kb, 16384, 128, 512, 1.0f);
  gemm_wmma_kernel<false><<<dim3(128, 4), blk, 0, stream>>>(
      kvbf, wvb, nullptr, Vb, 16384, 128, 512, 1.0f);
  attn_kernel<<<dim3(512), blk, 0, stream>>>(Qb, Kb, Vb, Ob);
  gemm_wmma_kernel<true><<<dim3(64, 4), blk, 0, stream>>>(
      Ob, wob, b_o, (float*)d_out, 8192, 512, 512, 1.0f);
}
